// TransformerMoEBlock_56066503082571
// MI455X (gfx1250) — matrix operand order repaired and checked by reading, not compiled, not run on hardware
//
#include <hip/hip_runtime.h>
#include <cstdint>
#include <cstddef>

// ---------------------------------------------------------------------------
// Problem constants (fixed by setup_inputs)
// ---------------------------------------------------------------------------
#define Bc 2
#define Sc 2048
#define Dc 1024
#define Pc 64
#define Ec 8
#define Ic 4096
#define Hc 16
#define Tc (Bc * Sc)   // 4096 tokens

typedef __attribute__((ext_vector_type(16))) __bf16 v16bf;
typedef __attribute__((ext_vector_type(8)))  float  f32x8;
typedef unsigned int u32x4 __attribute__((ext_vector_type(4)));
typedef int          i32x8 __attribute__((ext_vector_type(8)));
typedef int          i32x4 __attribute__((ext_vector_type(4)));

#if defined(__has_builtin)
#if __has_builtin(__builtin_amdgcn_tensor_load_to_lds) && \
    __has_builtin(__builtin_amdgcn_s_wait_tensorcnt)
#define USE_TDM 1
#endif
#endif
#ifndef USE_TDM
#define USE_TDM 0
#endif

union FragB { v16bf v; uint4 u[2]; };

__device__ __forceinline__ f32x8 zero8() {
  f32x8 z;
#pragma unroll
  for (int i = 0; i < 8; ++i) z[i] = 0.0f;
  return z;
}

__device__ __forceinline__ unsigned short bfbits(__bf16 x) {
  union { __bf16 b; unsigned short u; } c; c.b = x; return c.u;
}

// A-matrix fragment: lane holds row M=(lane&15); K elems {kb..kb+7, kb+16..kb+23}
// caller passes p = rowbase + kb where kb = (lane<16)?0:8
__device__ __forceinline__ v16bf frag_ld_a(const __bf16* p) {
  FragB f;
  f.u[0] = *reinterpret_cast<const uint4*>(p);
  f.u[1] = *reinterpret_cast<const uint4*>(p + 16);
  return f.v;
}

// B-matrix fragment: lane holds col N=(lane&15); K elems {kb..kb+7, kb+16..kb+23}
// caller passes p = rowbase + kb where kb = (lane<16)?0:8
__device__ __forceinline__ v16bf frag_ld_b(const __bf16* p) {
  FragB f;
  f.u[0] = *reinterpret_cast<const uint4*>(p);
  f.u[1] = *reinterpret_cast<const uint4*>(p + 16);
  return f.v;
}

__device__ __forceinline__ f32x8 wmma_bf16(v16bf a, v16bf b, f32x8 c) {
  return __builtin_amdgcn_wmma_f32_16x16x32_bf16(false, a, false, b, (short)0, c,
                                                 false, false);
}

// ---------------------------------------------------------------------------
// fp32 -> bf16 conversion (weights)
// ---------------------------------------------------------------------------
__global__ void f32_to_bf16_kernel(const float* __restrict__ in,
                                   __bf16* __restrict__ out, size_t n) {
  size_t i = (size_t)blockIdx.x * blockDim.x + threadIdx.x;
  size_t stride = (size_t)gridDim.x * blockDim.x;
  for (; i < n; i += stride) out[i] = (__bf16)in[i];
}

// ---------------------------------------------------------------------------
// mod = fluid @ aln_w + aln_b   (per batch, both LN sets)
// layout per batch b: [aln1 scale(D) | aln1 shift(D) | aln2 scale(D) | aln2 shift(D)]
// ---------------------------------------------------------------------------
__global__ __launch_bounds__(256) void mod_kernel(
    const float* __restrict__ fluid, const float* __restrict__ w1,
    const float* __restrict__ b1, const float* __restrict__ w2,
    const float* __restrict__ b2, float* __restrict__ mod) {
  const int b = blockIdx.y;
  const int j = blockIdx.x * 256 + threadIdx.x;  // 0..4D-1
  const int twoD = 2 * Dc;
  const float* w; const float* bias; int col;
  if (j < twoD) { w = w1; bias = b1; col = j; }
  else          { w = w2; bias = b2; col = j - twoD; }
  float acc = bias[col];
#pragma unroll 8
  for (int p = 0; p < Pc; ++p) acc += fluid[b * Pc + p] * w[(size_t)p * twoD + col];
  mod[(size_t)b * 4 * Dc + j] = acc;
}

// ---------------------------------------------------------------------------
// Adaptive layernorm -> bf16
// ---------------------------------------------------------------------------
__global__ __launch_bounds__(256) void adaln_kernel(
    const float* __restrict__ x, const float* __restrict__ mod, int which,
    __bf16* __restrict__ out) {
  __shared__ float ssum[256], ssq[256];
  const int t = blockIdx.x;
  const int b = t / Sc;
  const int tid = threadIdx.x;
  const float* xr = x + (size_t)t * Dc;
  const float* mr = mod + (size_t)b * 4 * Dc + which * 2 * Dc;
  float s = 0.f, q = 0.f;
  for (int d = tid; d < Dc; d += 256) { float v = xr[d]; s += v; q += v * v; }
  ssum[tid] = s; ssq[tid] = q;
  __syncthreads();
  for (int st = 128; st > 0; st >>= 1) {
    if (tid < st) { ssum[tid] += ssum[tid + st]; ssq[tid] += ssq[tid + st]; }
    __syncthreads();
  }
  const float mu = ssum[0] / Dc;
  const float var = ssq[0] / Dc - mu * mu;
  const float inv = rsqrtf(var + 1e-6f);
  for (int d = tid; d < Dc; d += 256) {
    float h = (xr[d] - mu) * inv;
    out[(size_t)t * Dc + d] = (__bf16)(h * (1.0f + mr[d]) + mr[Dc + d]);
  }
}

// ---------------------------------------------------------------------------
// Generic bf16 WMMA GEMM:  C(MxN) = A(MxK) * B(KxN)    (row-major)
// 128x128 block tile, 8 waves, 32x64 per-wave tile (8 WMMA / k-step).
// A tile staged by the Tensor Data Mover (TDM) with hardware LDS padding when
// available (pad_interval=16 DWORDs, pad_amount=8 DWORDs -> 48-elem rows).
// MODE 0: C_f32 = acc (+ resid)      MODE 1: C_bf16 = gelu(acc)
// MODE 2: C_f32 += rowscale[m*rs_stride] * acc
// Requires M%128==0, N%128==0, K%32==0.
// ---------------------------------------------------------------------------
#define GBM 128
#define GBN 128
#define GBK 32
#define PADA 48   // bf16 elems per LDS row (96B, 16B-aligned)
#define PADB 48
#define GEMM_LDS_BYTES ((GBM * PADA + GBN * PADB) * 2)

template <int MODE>
__global__ __launch_bounds__(256) void gemm_bf16_wmma(
    const __bf16* __restrict__ A, const __bf16* __restrict__ Bm,
    float* __restrict__ C, __bf16* __restrict__ Cbf,
    const float* __restrict__ resid, const float* __restrict__ rowscale,
    int rs_stride, int M, int N, int K) {
  extern __shared__ __align__(16) __bf16 dynsmem[];
  __bf16* sA  = dynsmem;                 // [128][PADA], group-segment offset 0
  __bf16* sBt = dynsmem + GBM * PADA;    // transposed: [n][k]

  const int tid = threadIdx.x;
  const int lane = tid & 31;
  const int wid = tid >> 5;   // 0..7
  const int wm = wid >> 1;    // 0..3 -> rows wm*32
  const int wn = wid & 1;     // 0..1 -> cols wn*64
  const int m0 = blockIdx.y * GBM;
  const int n0 = blockIdx.x * GBN;

  const int lrow = lane & 15;
  const int kbA = (lane < 16) ? 0 : 8;
  const int kbB = (lane < 16) ? 0 : 8;

  f32x8 acc[2][4];
#pragma unroll
  for (int i = 0; i < 2; ++i)
#pragma unroll
    for (int j = 0; j < 4; ++j) acc[i][j] = zero8();

#if !USE_TDM
  const int ar = tid >> 1;            // A stage: row 0..127
  const int ac = (tid & 1) * 16;      // A stage: col group (16 elems)
#endif
  const int bkp = tid >> 4;           // B stage: k pair 0..15
  const int bng = tid & 15;           // B stage: n group 0..15

  for (int k0 = 0; k0 < K; k0 += GBK) {
    // ---- stage A tile (128x32) -> sA with 48-elem padded rows ----
#if USE_TDM
    if (tid < 32) {
      const unsigned long long ga =
          (unsigned long long)(const void*)(A + (size_t)m0 * K + k0);
      u32x4 g0;
      g0[0] = 1u;                                    // count=1, user descriptor
      g0[1] = 0u;                                    // lds_addr = dyn smem base
      g0[2] = (unsigned)(ga & 0xFFFFFFFFull);        // global_addr lo
      g0[3] = (unsigned)((ga >> 32) & 0x01FFFFFFull) // global_addr hi (57b)
              | (2u << 30);                          // type = 2 (image)
      i32x8 g1;
      g1[0] = (int)((1u << 16)        // data_size = 2 bytes
                    | (1u << 20)      // pad_enable
                    | (3u << 22)      // pad_interval: 16 DWORDs (one 32-elem row)
                    | (7u << 25));    // pad_amount:  8 DWORDs (16 elems)
      g1[1] = (int)(((unsigned)K & 0xFFFFu) << 16);                  // dim0 lo16
      g1[2] = (int)(((unsigned)K >> 16) | (((unsigned)M & 0xFFFFu) << 16));
      g1[3] = (int)(((unsigned)M >> 16) | ((unsigned)GBK << 16));    // tile_dim0
      g1[4] = (int)GBM;                 // tile_dim1 = 128, tile_dim2 = 0
      g1[5] = (int)K;                   // tensor_dim0_stride lo32
      g1[6] = 0;
      g1[7] = 0;
      const i32x4 gz4 = {0, 0, 0, 0};
      const i32x8 gz8 = {0, 0, 0, 0, 0, 0, 0, 0};
      __builtin_amdgcn_tensor_load_to_lds(g0, g1, gz4, gz4, gz8, 0);
    }
#else
    {
      const __bf16* ga = A + (size_t)(m0 + ar) * K + k0 + ac;
      *reinterpret_cast<uint4*>(sA + ar * PADA + ac) =
          *reinterpret_cast<const uint4*>(ga);
      *reinterpret_cast<uint4*>(sA + ar * PADA + ac + 8) =
          *reinterpret_cast<const uint4*>(ga + 8);
    }
#endif
    // ---- stage B tile (32x128) transposed -> sBt[n][k] ----
    {
      const __bf16* g0 = Bm + (size_t)(k0 + 2 * bkp) * N + n0 + bng * 8;
      uint4 r0 = *reinterpret_cast<const uint4*>(g0);
      uint4 r1 = *reinterpret_cast<const uint4*>(g0 + N);
      if (k0 + GBK < K) __builtin_prefetch(g0 + (size_t)GBK * N, 0, 1);
      const __bf16* p0 = reinterpret_cast<const __bf16*>(&r0);
      const __bf16* p1 = reinterpret_cast<const __bf16*>(&r1);
#pragma unroll
      for (int j = 0; j < 8; ++j) {
        unsigned int v = (unsigned int)bfbits(p0[j]) |
                         ((unsigned int)bfbits(p1[j]) << 16);
        *reinterpret_cast<unsigned int*>(sBt + (bng * 8 + j) * PADB + 2 * bkp) = v;
      }
    }
#if USE_TDM
    if (tid < 32) __builtin_amdgcn_s_wait_tensorcnt(0);
#endif
    __syncthreads();

    v16bf af[2], bfv[4];
    af[0] = frag_ld_a(sA + (wm * 32 + lrow) * PADA + kbA);
    af[1] = frag_ld_a(sA + (wm * 32 + 16 + lrow) * PADA + kbA);
#pragma unroll
    for (int j = 0; j < 4; ++j)
      bfv[j] = frag_ld_b(sBt + (wn * 64 + j * 16 + lrow) * PADB + kbB);

#pragma unroll
    for (int i = 0; i < 2; ++i)
#pragma unroll
      for (int j = 0; j < 4; ++j)
        acc[i][j] = wmma_bf16(af[i], bfv[j], acc[i][j]);
    __syncthreads();
  }

  // ---- epilogue ----
  const int rbase = (lane >> 4) << 3;  // 0 or 8
#pragma unroll
  for (int i = 0; i < 2; ++i) {
#pragma unroll
    for (int j = 0; j < 4; ++j) {
#pragma unroll
      for (int r = 0; r < 8; ++r) {
        const int row = m0 + wm * 32 + i * 16 + rbase + r;
        const int col = n0 + wn * 64 + j * 16 + (lane & 15);
        const size_t off = (size_t)row * N + col;
        const float v = acc[i][j][r];
        if (MODE == 0) {
          C[off] = resid ? resid[off] + v : v;
        } else if (MODE == 1) {
          const float g = 0.5f * v * (1.0f + erff(v * 0.7071067811865475f));
          Cbf[off] = (__bf16)g;
        } else {
          C[off] += rowscale[(size_t)row * rs_stride] * v;
        }
      }
    }
  }
}

// ---------------------------------------------------------------------------
// RoPE + split qkv(f32, [t][3D]) -> q,k (roped) and v, bf16 [b][h][s][hd]
// ---------------------------------------------------------------------------
__global__ void rope_kernel(const float* __restrict__ qkv,
                            const float* __restrict__ freqs,
                            __bf16* __restrict__ q, __bf16* __restrict__ k,
                            __bf16* __restrict__ v) {
  size_t i = (size_t)blockIdx.x * 256 + threadIdx.x;  // pair index
  const int p = i & 31; i >>= 5;
  const int h = i & 15; i >>= 4;
  const size_t t = i;            // 0..Tc-1
  const int s = (int)(t % Sc);
  const int b = (int)(t / Sc);

  const float f = freqs[(size_t)s * 32 + p];
  const float cs = cosf(f), sn = sinf(f);

  const float* base = qkv + t * (3 * Dc) + h * 64;
  const float q1 = base[2 * p],          q2 = base[2 * p + 1];
  const float k1 = base[Dc + 2 * p],     k2 = base[Dc + 2 * p + 1];
  const float v1 = base[2 * Dc + 2 * p], v2 = base[2 * Dc + 2 * p + 1];

  const size_t ho = (((size_t)(b * Hc + h)) * Sc + s) * 64;
  q[ho + 2 * p]     = (__bf16)(q1 * cs - q2 * sn);
  q[ho + 2 * p + 1] = (__bf16)(q1 * sn + q2 * cs);
  k[ho + 2 * p]     = (__bf16)(k1 * cs - k2 * sn);
  k[ho + 2 * p + 1] = (__bf16)(k1 * sn + k2 * cs);
  v[ho + 2 * p]     = (__bf16)v1;
  v[ho + 2 * p + 1] = (__bf16)v2;
}

// ---------------------------------------------------------------------------
// Flash attention, WMMA bf16. 4 waves/block; wave owns 16 Q rows; 32-key tiles.
// Q,K,V: [b][h][s][64] bf16.  O: [b][s][h*64] bf16 (= [t][D]).
// ---------------------------------------------------------------------------
#define KSTR 80   // sK row stride (bf16 elems, 160B)
#define VSTR 48   // sVt / sP row stride (96B)

__global__ __launch_bounds__(128) void attn_kernel(
    const __bf16* __restrict__ Q, const __bf16* __restrict__ Kt,
    const __bf16* __restrict__ V, __bf16* __restrict__ O) {
  __shared__ __align__(16) __bf16 sK[32 * KSTR];   // [key][hd]
  __shared__ __align__(16) __bf16 sVt[64 * VSTR];  // [hd][key]
  __shared__ __align__(16) __bf16 sP[4 * 16 * VSTR];

  const int b = blockIdx.z, h = blockIdx.y;
  const int q0 = blockIdx.x * 64;
  const int tid = threadIdx.x, lane = tid & 31, wv = tid >> 5;
  const int lrow = lane & 15;
  const int kbA = (lane < 16) ? 0 : 8;
  const int kbB = (lane < 16) ? 0 : 8;
  const size_t headoff = ((size_t)(b * Hc + h)) * Sc * 64;

  // preload this wave's Q fragments (hd chunks 0..31, 32..63)
  const __bf16* qrow = Q + headoff + (size_t)(q0 + wv * 16 + lrow) * 64;
  const v16bf qf0 = frag_ld_a(qrow + kbA);
  const v16bf qf1 = frag_ld_a(qrow + 32 + kbA);

  f32x8 oacc[4];
#pragma unroll
  for (int j = 0; j < 4; ++j) oacc[j] = zero8();
  float mrow[8], lsum[8];
#pragma unroll
  for (int r = 0; r < 8; ++r) { mrow[r] = -1e30f; lsum[r] = 0.0f; }

  const int skey = tid >> 3;           // K stage key 0..15 (+16)
  const int scol = (tid & 7) * 8;      // K stage hd group
  const int vkp = tid >> 3;            // V stage key pair 0..15
  const int vhg = (tid & 7) * 8;       // V stage hd group

  for (int kt = 0; kt < Sc; kt += 32) {
    // ---- stage K (32x64, natural) and V (transposed [hd][key]) ----
    *reinterpret_cast<uint4*>(sK + skey * KSTR + scol) =
        *reinterpret_cast<const uint4*>(Kt + headoff + (size_t)(kt + skey) * 64 + scol);
    *reinterpret_cast<uint4*>(sK + (skey + 16) * KSTR + scol) =
        *reinterpret_cast<const uint4*>(Kt + headoff + (size_t)(kt + skey + 16) * 64 + scol);
    {
      const __bf16* g0 = V + headoff + (size_t)(kt + 2 * vkp) * 64 + vhg;
      uint4 r0 = *reinterpret_cast<const uint4*>(g0);
      uint4 r1 = *reinterpret_cast<const uint4*>(g0 + 64);
      const __bf16* p0 = reinterpret_cast<const __bf16*>(&r0);
      const __bf16* p1 = reinterpret_cast<const __bf16*>(&r1);
#pragma unroll
      for (int j = 0; j < 8; ++j) {
        unsigned int vv = (unsigned int)bfbits(p0[j]) |
                          ((unsigned int)bfbits(p1[j]) << 16);
        *reinterpret_cast<unsigned int*>(sVt + (vhg + j) * VSTR + 2 * vkp) = vv;
      }
    }
    __syncthreads();

    // ---- scores: S(16x32) = Q(16x64) . K_tile^T ----
    f32x8 s0 = zero8(), s1 = zero8();
    s0 = wmma_bf16(qf0, frag_ld_b(sK + lrow * KSTR + kbB), s0);
    s0 = wmma_bf16(qf1, frag_ld_b(sK + lrow * KSTR + 32 + kbB), s0);
    s1 = wmma_bf16(qf0, frag_ld_b(sK + (16 + lrow) * KSTR + kbB), s1);
    s1 = wmma_bf16(qf1, frag_ld_b(sK + (16 + lrow) * KSTR + 32 + kbB), s1);

    // ---- online softmax update (row stats across 16-lane groups) ----
    __bf16* pbase = sP + wv * 16 * VSTR;
#pragma unroll
    for (int r = 0; r < 8; ++r) {
      float a = s0[r] * 0.125f;
      float c = s1[r] * 0.125f;
      float mx = fmaxf(a, c);
#pragma unroll
      for (int d = 1; d < 16; d <<= 1) mx = fmaxf(mx, __shfl_xor(mx, d, 32));
      const float mnew = fmaxf(mrow[r], mx);
      const float scale = __expf(mrow[r] - mnew);
      const float p0v = __expf(a - mnew);
      const float p1v = __expf(c - mnew);
      float sum = p0v + p1v;
#pragma unroll
      for (int d = 1; d < 16; d <<= 1) sum += __shfl_xor(sum, d, 32);
      lsum[r] = lsum[r] * scale + sum;
      mrow[r] = mnew;
#pragma unroll
      for (int j = 0; j < 4; ++j) oacc[j][r] *= scale;
      __bf16* pp = pbase + (r + ((lane >> 4) << 3)) * VSTR + (lane & 15);
      pp[0]  = (__bf16)p0v;
      pp[16] = (__bf16)p1v;
    }
    __syncthreads();

    // ---- O += P(16x32) . V(32x64) ----
    const v16bf pf = frag_ld_a(pbase + lrow * VSTR + kbA);
#pragma unroll
    for (int j = 0; j < 4; ++j)
      oacc[j] = wmma_bf16(pf, frag_ld_b(sVt + (j * 16 + lrow) * VSTR + kbB), oacc[j]);
    __syncthreads();
  }

  // ---- normalize + store O [b][s][h*64+..] bf16 ----
#pragma unroll
  for (int r = 0; r < 8; ++r) {
    const float inv = 1.0f / lsum[r];
    const int srow = q0 + wv * 16 + r + ((lane >> 4) << 3);
#pragma unroll
    for (int j = 0; j < 4; ++j) {
      const int col = h * 64 + j * 16 + (lane & 15);
      O[((size_t)(b * Sc + srow)) * Dc + col] = (__bf16)(oacc[j][r] * inv);
    }
  }
}

// ---------------------------------------------------------------------------
// Router: logits = xn2 @ w_router; top-k on logits+b; softmax of gathered logits
// ---------------------------------------------------------------------------
__global__ __launch_bounds__(128) void router_kernel(
    const __bf16* __restrict__ xn, const float* __restrict__ wr,
    const float* __restrict__ br, const int* __restrict__ topk_p,
    float* __restrict__ gates) {
  __shared__ float red[128][Ec];
  const int t = blockIdx.x, tid = threadIdx.x;
  float acc[Ec];
#pragma unroll
  for (int e = 0; e < Ec; ++e) acc[e] = 0.0f;
  for (int d = tid; d < Dc; d += 128) {
    const float xv = (float)xn[(size_t)t * Dc + d];
    const float* w = wr + (size_t)d * Ec;
#pragma unroll
    for (int e = 0; e < Ec; ++e) acc[e] += xv * w[e];
  }
#pragma unroll
  for (int e = 0; e < Ec; ++e) red[tid][e] = acc[e];
  __syncthreads();
  for (int s = 64; s > 0; s >>= 1) {
    if (tid < s)
#pragma unroll
      for (int e = 0; e < Ec; ++e) red[tid][e] += red[tid + s][e];
    __syncthreads();
  }
  if (tid == 0) {
    float logits[Ec], sel[Ec];
    bool used[Ec];
#pragma unroll
    for (int e = 0; e < Ec; ++e) {
      logits[e] = red[0][e];
      sel[e] = logits[e] + br[e];
      used[e] = false;
      gates[(size_t)t * Ec + e] = 0.0f;
    }
    int tk = *topk_p;
    if (tk > Ec) tk = Ec;
    int idx[Ec]; float gl[Ec];
    for (int k = 0; k < tk; ++k) {
      int best = 0; float bv = -1e30f;
      for (int e = 0; e < Ec; ++e)
        if (!used[e] && sel[e] > bv) { bv = sel[e]; best = e; }
      used[best] = true; idx[k] = best; gl[k] = logits[best];
    }
    float mx = -1e30f;
    for (int k = 0; k < tk; ++k) mx = fmaxf(mx, gl[k]);
    float s = 0.0f;
    for (int k = 0; k < tk; ++k) { gl[k] = __expf(gl[k] - mx); s += gl[k]; }
    for (int k = 0; k < tk; ++k) gates[(size_t)t * Ec + idx[k]] = gl[k] / s;
  }
}

// ---------------------------------------------------------------------------
// out = a + b
// ---------------------------------------------------------------------------
__global__ void add_kernel(const float* __restrict__ a,
                           const float* __restrict__ b,
                           float* __restrict__ o, size_t n) {
  size_t i = (size_t)blockIdx.x * blockDim.x + threadIdx.x;
  size_t stride = (size_t)gridDim.x * blockDim.x;
  for (; i < n; i += stride) o[i] = a[i] + b[i];
}

// ---------------------------------------------------------------------------
// Host launcher
// ---------------------------------------------------------------------------
extern "C" void kernel_launch(void* const* d_in, const int* in_sizes, int n_in,
                              void* d_out, int out_size, void* d_ws,
                              size_t ws_size, hipStream_t stream) {
  const float* x        = (const float*)d_in[0];
  const float* freqs    = (const float*)d_in[1];
  const float* fluid    = (const float*)d_in[2];
  const float* aln1_w   = (const float*)d_in[3];
  const float* aln1_b   = (const float*)d_in[4];
  const float* aln2_w   = (const float*)d_in[5];
  const float* aln2_b   = (const float*)d_in[6];
  const float* wqkv     = (const float*)d_in[7];
  const float* wo       = (const float*)d_in[8];
  const float* w_router = (const float*)d_in[9];
  const float* b_router = (const float*)d_in[10];
  const float* w1       = (const float*)d_in[11];
  const float* w2       = (const float*)d_in[12];
  const int*   topk     = (const int*)d_in[13];
  float* out = (float*)d_out;

  char* ws = (char*)d_ws;
  size_t off = 0;
  auto take = [&](size_t bytes) -> char* {
    char* p = ws + off;
    off += (bytes + 255) & ~(size_t)255;
    return p;
  };

  __bf16* wqkv_b = (__bf16*)take((size_t)Dc * 3 * Dc * 2);
  __bf16* wo_b   = (__bf16*)take((size_t)Dc * Dc * 2);
  __bf16* w1_b   = (__bf16*)take((size_t)Ec * Dc * Ic * 2);
  __bf16* w2_b   = (__bf16*)take((size_t)Ec * Ic * Dc * 2);
  float*  modb   = (float*)take((size_t)Bc * 4 * Dc * 4);
  __bf16* xn     = (__bf16*)take((size_t)Tc * Dc * 2);
  float*  qkv    = (float*)take((size_t)Tc * 3 * Dc * 4);
  __bf16* qb     = (__bf16*)take((size_t)Tc * Dc * 2);
  __bf16* kb     = (__bf16*)take((size_t)Tc * Dc * 2);
  __bf16* vb     = (__bf16*)take((size_t)Tc * Dc * 2);
  __bf16* ob     = (__bf16*)take((size_t)Tc * Dc * 2);
  float*  x1     = (float*)take((size_t)Tc * Dc * 4);
  float*  gates  = (float*)take((size_t)Tc * Ec * 4);
  __bf16* hbuf   = (__bf16*)take((size_t)Tc * Ic * 2);
  float*  ybuf   = (float*)take((size_t)Tc * Dc * 4);

  // weights -> bf16
  f32_to_bf16_kernel<<<4096, 256, 0, stream>>>(wqkv, wqkv_b, (size_t)Dc * 3 * Dc);
  f32_to_bf16_kernel<<<4096, 256, 0, stream>>>(wo, wo_b, (size_t)Dc * Dc);
  f32_to_bf16_kernel<<<8192, 256, 0, stream>>>(w1, w1_b, (size_t)Ec * Dc * Ic);
  f32_to_bf16_kernel<<<8192, 256, 0, stream>>>(w2, w2_b, (size_t)Ec * Ic * Dc);

  // adaptive LN modulation + LN1
  mod_kernel<<<dim3(4 * Dc / 256, Bc), 256, 0, stream>>>(fluid, aln1_w, aln1_b,
                                                         aln2_w, aln2_b, modb);
  adaln_kernel<<<Tc, 256, 0, stream>>>(x, modb, 0, xn);

  // qkv projection
  gemm_bf16_wmma<0><<<dim3(3 * Dc / GBN, Tc / GBM), 256, GEMM_LDS_BYTES, stream>>>(
      xn, wqkv_b, qkv, nullptr, nullptr, nullptr, 0, Tc, 3 * Dc, Dc);

  // rope + split
  rope_kernel<<<(Tc * Hc * 32) / 256, 256, 0, stream>>>(qkv, freqs, qb, kb, vb);

  // flash attention
  attn_kernel<<<dim3(Sc / 64, Hc, Bc), 128, 0, stream>>>(qb, kb, vb, ob);

  // out projection + residual
  gemm_bf16_wmma<0><<<dim3(Dc / GBN, Tc / GBM), 256, GEMM_LDS_BYTES, stream>>>(
      ob, wo_b, x1, nullptr, x, nullptr, 0, Tc, Dc, Dc);

  // LN2 + router
  adaln_kernel<<<Tc, 256, 0, stream>>>(x1, modb, 1, xn);
  router_kernel<<<Tc, 128, 0, stream>>>(xn, w_router, b_router, topk, gates);

  // MoE experts (dense, matching the reference loop)
  (void)hipMemsetAsync(ybuf, 0, (size_t)Tc * Dc * 4, stream);
  for (int e = 0; e < Ec; ++e) {
    gemm_bf16_wmma<1><<<dim3(Ic / GBN, Tc / GBM), 256, GEMM_LDS_BYTES, stream>>>(
        xn, w1_b + (size_t)e * Dc * Ic, nullptr, hbuf, nullptr, nullptr, 0, Tc,
        Ic, Dc);
    gemm_bf16_wmma<2><<<dim3(Dc / GBN, Tc / GBM), 256, GEMM_LDS_BYTES, stream>>>(
        hbuf, w2_b + (size_t)e * Ic * Dc, ybuf, nullptr, nullptr, gates + e, Ec,
        Tc, Dc, Ic);
  }

  // final residual
  add_kernel<<<4096, 256, 0, stream>>>(x1, ybuf, out, (size_t)Tc * Dc);
}
